// BMM_S8T_S8N_F16T_72292889526616
// MI455X (gfx1250) — compile-verified, matrix operand order checked by reading, not run on hardware
//
#include <hip/hip_runtime.h>

// BMM_S8T_S8N_F16T: out[b,m,n] = fp16( alpha * sum_k a[b,m,k]*b[b,n,k] )
// B=64, M=N=1024, K=128. int8 math via V_WMMA_I32_16X16X64_IU8 (wave32).
//
// Roofline (MI455X, 23.3 TB/s HBM): pack pass 64MB rd(NT) + 16MB wr,
// GEMM pass 16MB rd (L2-resident re-reads) + 128MB fp16 wr(NT) -> memory
// bound, ~12us; IU8 WMMA compute is negligible. NT hints keep the 16MB
// packed operands resident in L2 while the streaming data bypasses it.

#define BB 64
#define MM 1024
#define NN 1024
#define KK 128

typedef __attribute__((ext_vector_type(8))) int v8i;
typedef __attribute__((ext_vector_type(4))) int v4i_t;

// ---------------------------------------------------------------------------
// Pass 1: pack int32 (int8-range values) -> int8 bytes in workspace.
// Each thread packs 4 consecutive elements (16B NT load -> 4B store).
// Source is never re-read: non-temporal loads. Packed result IS re-read by
// the GEMM pass: regular-temporal stores (stay in L2).
// ---------------------------------------------------------------------------
__global__ __launch_bounds__(256) void pack_s8_kernel(const int* __restrict__ src,
                                                      unsigned* __restrict__ dst) {
    const int idx = blockIdx.x * blockDim.x + threadIdx.x;
    const v4i_t v = __builtin_nontemporal_load((const v4i_t*)src + idx);
    unsigned p = (unsigned)(v.x & 0xFF)
               | ((unsigned)(v.y & 0xFF) << 8)
               | ((unsigned)(v.z & 0xFF) << 16)
               | ((unsigned)(v.w & 0xFF) << 24);
    dst[idx] = p;
}

// ---------------------------------------------------------------------------
// Load a 16x64 int8 operand fragment (row-major, row stride = KK bytes) into
// the ISA 8-bit A/B-operand VGPR layout (CDNA5 ISA 7.12.2):
//   lanes 0-15 : rows 0-15, K chunks {0-7,16-23,32-39,48-55}
//   lanes 16-31: rows 0-15, K chunks {8-15,24-31,40-47,56-63}
// Each lane: four 8-byte loads at k = 16*j + 8*(lane>=16).
// B is stored N-major/K-contiguous (b[b,n,k]) so it uses the same loader.
// ---------------------------------------------------------------------------
__device__ inline v8i load_frag_s8(const signed char* __restrict__ base, int lane) {
    const int r = lane & 15;
    const int h = lane >> 4;
    const char* p = (const char*)base + r * KK + h * 8;
    int2 d0 = *(const int2*)(p +  0);
    int2 d1 = *(const int2*)(p + 16);
    int2 d2 = *(const int2*)(p + 32);
    int2 d3 = *(const int2*)(p + 48);
    v8i f;
    f[0] = d0.x; f[1] = d0.y;
    f[2] = d1.x; f[3] = d1.y;
    f[4] = d2.x; f[5] = d2.y;
    f[6] = d3.x; f[7] = d3.y;
    return f;
}

// ---------------------------------------------------------------------------
// Pass 2: WMMA GEMM. Block = 256 threads = 8 wave32s, block tile 128x128,
// wave tile 64x32 (4x2 tiles of 16x16). K=128 -> two k=64 WMMA steps.
// ---------------------------------------------------------------------------
__global__ __launch_bounds__(256) void bmm_s8_wmma_kernel(
    const signed char* __restrict__ aPack,   // [B, M, K] int8
    const signed char* __restrict__ bPack,   // [B, N, K] int8
    const float* __restrict__ alphaPtr,
    _Float16* __restrict__ out)              // [B, M, N] fp16
{
    const int lane = threadIdx.x & 31;
    const int wave = threadIdx.x >> 5;
    const int wm   = wave & 1;    // 2 waves along M
    const int wn   = wave >> 1;   // 4 waves along N
    const int bz   = blockIdx.z;
    const int M0   = blockIdx.y * 128 + wm * 64;
    const int N0   = blockIdx.x * 128 + wn * 32;

    const signed char* aBase = aPack + ((size_t)bz * MM + M0) * KK;
    const signed char* bBase = bPack + ((size_t)bz * NN + N0) * KK;

    v8i acc[4][2] = {};   // 4 M-tiles x 2 N-tiles, i32 16x16 each (8 VGPRs)

    #pragma unroll
    for (int kk = 0; kk < KK; kk += 64) {
        v8i afrag[4];
        v8i bfrag[2];
        #pragma unroll
        for (int i = 0; i < 4; ++i)
            afrag[i] = load_frag_s8(aBase + (size_t)(16 * i) * KK + kk, lane);
        #pragma unroll
        for (int j = 0; j < 2; ++j)
            bfrag[j] = load_frag_s8(bBase + (size_t)(16 * j) * KK + kk, lane);

        #pragma unroll
        for (int i = 0; i < 4; ++i)
            #pragma unroll
            for (int j = 0; j < 2; ++j)
                // (sgn_a, A, sgn_b, B, C, reuse_a, reuse_b)
                acc[i][j] = __builtin_amdgcn_wmma_i32_16x16x64_iu8(
                    true, afrag[i], true, bfrag[j], acc[i][j], false, false);
    }

    // Epilogue: i32 acc -> alpha * f32 -> fp16, non-temporal store (write-once
    // stream; keep it out of L2 so packed operands stay resident).
    // C/D layout: VGPR r: lanes 0-15 -> M=r, N=lane; lanes 16-31 -> M=r+8, N=lane-16.
    const float alpha = *alphaPtr;
    const int h = lane >> 4;
    const int c = lane & 15;
    #pragma unroll
    for (int i = 0; i < 4; ++i) {
        #pragma unroll
        for (int j = 0; j < 2; ++j) {
            size_t base = ((size_t)bz * MM + (size_t)(M0 + 16 * i + 8 * h)) * NN
                        + (size_t)(N0 + 16 * j + c);
            #pragma unroll
            for (int r = 0; r < 8; ++r) {
                _Float16 val = (_Float16)(alpha * (float)acc[i][j][r]);
                __builtin_nontemporal_store(val, &out[base + (size_t)r * NN]);
            }
        }
    }
}

// ---------------------------------------------------------------------------
// Host-side launcher (graph-capture safe: only kernel launches on stream).
// d_ws usage: A packed (8 MB) followed by B packed (8 MB) = 16 MB.
// ---------------------------------------------------------------------------
extern "C" void kernel_launch(void* const* d_in, const int* in_sizes, int n_in,
                              void* d_out, int out_size, void* d_ws, size_t ws_size,
                              hipStream_t stream) {
    const int*   a     = (const int*)d_in[0];    // [B, M, K] int32
    const int*   b     = (const int*)d_in[1];    // [B, N, K] int32
    const float* alpha = (const float*)d_in[2];  // [1] float
    _Float16*    out   = (_Float16*)d_out;       // [B, M, N] fp16

    signed char* aPack = (signed char*)d_ws;
    signed char* bPack = aPack + (size_t)BB * MM * KK;

    const int elems  = BB * MM * KK;             // 8,388,608 per matrix
    const int blocks = elems / 4 / 256;          // 4 elems per thread

    pack_s8_kernel<<<blocks, 256, 0, stream>>>(a, (unsigned*)aPack);
    pack_s8_kernel<<<blocks, 256, 0, stream>>>(b, (unsigned*)bPack);

    dim3 grid(NN / 128, MM / 128, BB);
    bmm_s8_wmma_kernel<<<grid, 256, 0, stream>>>(aPack, bPack, alpha, out);
}
